// EncLayer_87333864997632
// MI455X (gfx1250) — matrix operand order repaired and checked by reading, not compiled, not run on hardware
//
#include <hip/hip_runtime.h>
#include <hip/hip_bf16.h>

typedef __attribute__((ext_vector_type(16))) __bf16 v16bf;
typedef __attribute__((ext_vector_type(8)))  float  v8f;

union FragBF {
    v16bf v;
    unsigned int u[8];
};

__device__ __forceinline__ unsigned short f2bf(float f) {
    unsigned int u = __float_as_uint(f);
    u += 0x7FFFu + ((u >> 16) & 1u);
    return (unsigned short)(u >> 16);
}
__device__ __forceinline__ unsigned int pack2bf(float a, float b) {
    return (unsigned int)f2bf(a) | ((unsigned int)f2bf(b) << 16);
}
__device__ __forceinline__ v8f wmma_bf16(const FragBF& a, const FragBF& b, v8f c) {
    return __builtin_amdgcn_wmma_f32_16x16x32_bf16(false, a.v, false, b.v, (short)0, c, false, false);
}
__device__ __forceinline__ void frag_from(FragBF& f, uint4 w0, uint4 w1) {
    f.u[0] = w0.x; f.u[1] = w0.y; f.u[2] = w0.z; f.u[3] = w0.w;
    f.u[4] = w1.x; f.u[5] = w1.y; f.u[6] = w1.z; f.u[7] = w1.w;
}

// ---------------- workspace layout (bytes) ----------------
#define OFF_Y      ((size_t)0)          // 16*4096*128 f32  = 33554432
#define OFF_CWBF   ((size_t)33554432)   // conv_w bf16 row-major 128x512       = 131072
#define OFF_CWA    ((size_t)33685504)   // codewords frag words k*64+d2 u32    = 4096
#define OFF_C2     ((size_t)33689600)   // 16 f32 (pad 256)
#define OFF_LWB    ((size_t)33689856)   // lin_w frag words c*1024+dk2 u32     = 2097152
#define OFF_STATS  ((size_t)35787008)   // 256 f32
#define OFF_BNP    ((size_t)35788032)   // 256 f32 (sa,sb)
#define OFF_E      ((size_t)35789056)   // 16*2048 f32 = 131072
#define OFF_ASUM   ((size_t)35920128)   // 16*16 f32 (pad 1024)
#define OFF_EA     ((size_t)35921152)   // e frag words b*1024+dk2 u32 = 65536
#define OFF_GATE   ((size_t)35986688)   // 16*512 f32 = 32768

// ================= K0: repack weights to bf16 fragment layouts, zero accumulators =================
__global__ __launch_bounds__(256) void k0_prep(
    const float* __restrict__ lin_w, const float* __restrict__ conv_w,
    const float* __restrict__ codewords,
    unsigned int* __restrict__ lwB, unsigned short* __restrict__ cwbf,
    unsigned int* __restrict__ cwA, float* __restrict__ c2,
    float* __restrict__ stats, float* __restrict__ E, float* __restrict__ asum)
{
    const int tid = blockIdx.x * 256 + threadIdx.x;
    if (tid < 524288) {                       // lin_w (512 x 2048) -> words c*1024 + dk2
        int c = tid >> 10, dk2 = tid & 1023;
        float a  = lin_w[(size_t)c * 2048 + 2 * dk2];
        float b2 = lin_w[(size_t)c * 2048 + 2 * dk2 + 1];
        lwB[tid] = pack2bf(a, b2);            // c*1024 + dk2 == tid
    } else if (tid < 589824) {                // conv_w f32 -> bf16 row major (128x512)
        int j = tid - 524288;
        cwbf[j] = f2bf(conv_w[j]);
    } else if (tid < 590848) {                // codewords (16x128) -> words k*64 + d2
        int j = tid - 589824, k = j >> 6, d2 = j & 63;
        cwA[j] = pack2bf(codewords[k * 128 + 2 * d2], codewords[k * 128 + 2 * d2 + 1]);
    } else if (tid < 590864) {                // c2[k] = ||codeword_k||^2
        int k = tid - 590848;
        float s = 0.f;
        for (int d = 0; d < 128; ++d) { float cv = codewords[k * 128 + d]; s = fmaf(cv, cv, s); }
        c2[k] = s;
    } else if (tid < 591120) {
        stats[tid - 590864] = 0.f;
    } else if (tid < 623888) {
        E[tid - 591120] = 0.f;
    } else if (tid < 624144) {
        asum[tid - 623888] = 0.f;
    }
}

// ================= K1: y = conv_w @ x + b (per batch 128x512 @ 512x4096) + BN stats ==============
// grid = 16*64 blocks, 256 threads (8 waves). Wave w owns d rows [16w,16w+16), block owns 64 tokens.
// LDS x tile: token-major, word = n*36 + k2  (k2 = local k / 2), 2304 words.
__global__ __launch_bounds__(256) void k1_conv(
    const float* __restrict__ x, const unsigned short* __restrict__ cwbf,
    const float* __restrict__ conv_b, float* __restrict__ y, float* __restrict__ stats)
{
    __shared__ unsigned int xsw[64 * 36];
    const int tid = threadIdx.x, wave = tid >> 5, lane = tid & 31;
    const int lh = lane & 15, hi = lane >> 4;
    const int b = blockIdx.x >> 6, n0 = (blockIdx.x & 63) * 64;
    const int d0 = wave * 16;
    v8f acc[4];
    acc[0] = {}; acc[1] = {}; acc[2] = {}; acc[3] = {};
    const int k2row = tid >> 3, oct = tid & 7;         // staging: 2 k-rows, 8 tokens

    for (int kc = 0; kc < 512; kc += 64) {
        const float* r0 = x + ((size_t)(b * 512 + kc + 2 * k2row)) * 4096 + n0 + oct * 8;
        const float* r1 = r0 + 4096;
        if (kc + 64 < 512) __builtin_prefetch(r0 + 64 * 4096, 0, 0);
        float4 a0 = ((const float4*)r0)[0], a1 = ((const float4*)r0)[1];
        float4 b0 = ((const float4*)r1)[0], b1 = ((const float4*)r1)[1];
        {
            int nb = oct * 8;
            xsw[(nb + 0) * 36 + k2row] = pack2bf(a0.x, b0.x);
            xsw[(nb + 1) * 36 + k2row] = pack2bf(a0.y, b0.y);
            xsw[(nb + 2) * 36 + k2row] = pack2bf(a0.z, b0.z);
            xsw[(nb + 3) * 36 + k2row] = pack2bf(a0.w, b0.w);
            xsw[(nb + 4) * 36 + k2row] = pack2bf(a1.x, b1.x);
            xsw[(nb + 5) * 36 + k2row] = pack2bf(a1.y, b1.y);
            xsw[(nb + 6) * 36 + k2row] = pack2bf(a1.z, b1.z);
            xsw[(nb + 7) * 36 + k2row] = pack2bf(a1.w, b1.w);
        }
        __syncthreads();
        #pragma unroll
        for (int kss = 0; kss < 2; ++kss) {
            const int ks = kss * 32;
            FragBF a;
            const uint4* ap = (const uint4*)(cwbf + (size_t)(d0 + lh) * 512 + kc + ks + hi * 8);
            frag_from(a, ap[0], ap[2]);                // K octets at +0 and +16 bf16
            #pragma unroll
            for (int t = 0; t < 4; ++t) {
                FragBF bf;
                const uint4* bp = (const uint4*)(xsw + (t * 16 + lh) * 36 + (ks >> 1) + hi * 4);
                frag_from(bf, bp[0], bp[2]);
                acc[t] = wmma_bf16(a, bf, acc[t]);
            }
        }
        __syncthreads();
    }
    float bias[8];
    #pragma unroll
    for (int r = 0; r < 8; ++r) bias[r] = conv_b[d0 + 8 * hi + r];
    #pragma unroll
    for (int t = 0; t < 4; ++t) {
        #pragma unroll
        for (int r = 0; r < 8; ++r) acc[t][r] += bias[r];
        float* dst = y + ((size_t)b * 4096 + n0 + t * 16 + lh) * 128 + d0 + 8 * hi;
        float4 lo4 = {acc[t][0], acc[t][1], acc[t][2], acc[t][3]};
        float4 hi4 = {acc[t][4], acc[t][5], acc[t][6], acc[t][7]};
        ((float4*)dst)[0] = lo4; ((float4*)dst)[1] = hi4;
    }
    #pragma unroll
    for (int r = 0; r < 8; ++r) {
        float s = acc[0][r] + acc[1][r] + acc[2][r] + acc[3][r];
        float q = acc[0][r]*acc[0][r] + acc[1][r]*acc[1][r] + acc[2][r]*acc[2][r] + acc[3][r]*acc[3][r];
        s += __shfl_xor(s, 1); s += __shfl_xor(s, 2); s += __shfl_xor(s, 4); s += __shfl_xor(s, 8);
        q += __shfl_xor(q, 1); q += __shfl_xor(q, 2); q += __shfl_xor(q, 4); q += __shfl_xor(q, 8);
        if (lh == 0) {
            atomicAdd(&stats[d0 + 8 * hi + r], s);
            atomicAdd(&stats[128 + d0 + 8 * hi + r], q);
        }
    }
}

// ================= K2: BN params ==================================================================
__global__ __launch_bounds__(128) void k2_bn(
    const float* __restrict__ stats, const float* __restrict__ bn_g,
    const float* __restrict__ bn_b, float* __restrict__ bnp)
{
    int t = threadIdx.x;
    if (t < 128) {
        float mean = stats[t] * (1.0f / 65536.0f);
        float var  = stats[128 + t] * (1.0f / 65536.0f) - mean * mean;
        float sa = bn_g[t] * rsqrtf(var + 1e-5f);
        bnp[t] = sa;
        bnp[128 + t] = bn_b[t] - mean * sa;
    }
}

// ================= K3: encoding (distances+softmax+aggregation) ==================================
// grid = 16b * 32slices, block 64 (2 waves). Each wave: 2 chunks of 32 tokens, private LDS slab:
//   yw: 32 tok x 68 words (X, d-paired)     [0,2176)
//   yt: 128 d  x 20 words (X^T, tok-paired) [2176,4736)
//   aw: 16 k   x 20 words (A,   tok-paired) [4736,5056)
__global__ __launch_bounds__(64) void k3_encode(
    const float* __restrict__ y, const float* __restrict__ bnp,
    const unsigned int* __restrict__ cwA, const float* __restrict__ c2ws,
    const float* __restrict__ scale, float* __restrict__ Ews, float* __restrict__ asumws)
{
    __shared__ unsigned int lds[2 * 5056];
    const int tid = threadIdx.x, wave = tid >> 5, lane = tid & 31;
    const int lh = lane & 15, hi = lane >> 4;
    const int b = blockIdx.x >> 5, slice = blockIdx.x & 31;
    const int g = slice * 2 + wave;
    unsigned int* wl = lds + wave * 5056;
    const int ytb = 2176, awb = 4736;
    unsigned short* l16 = (unsigned short*)wl;

    float sc[8], cc[8];
    #pragma unroll
    for (int r = 0; r < 8; ++r) { sc[r] = scale[8 * hi + r]; cc[r] = c2ws[8 * hi + r]; }

    v8f Eacc[8];
    #pragma unroll
    for (int s = 0; s < 8; ++s) Eacc[s] = {};
    float asum[8] = {0.f, 0.f, 0.f, 0.f, 0.f, 0.f, 0.f, 0.f};

    for (int chunk = 0; chunk < 2; ++chunk) {
        const int nbase = g * 64 + chunk * 32;
        // --- stage yv = relu(bn(y)) for token (nbase+lane): yw (u32 d-pairs) + yt (u16 scatter) ---
        const float* yp = y + ((size_t)b * 4096 + nbase + lane) * 128;
        if (chunk == 0) __builtin_prefetch(yp + 32 * 128, 0, 0);
        const int tw = lane >> 1, tp = lane & 1;
        float x2 = 0.f;
        #pragma unroll
        for (int i = 0; i < 32; ++i) {
            float4 v  = ((const float4*)yp)[i];
            float4 sa = ((const float4*)bnp)[i];
            float4 sb = ((const float4*)(bnp + 128))[i];
            float v0 = fmaxf(fmaf(v.x, sa.x, sb.x), 0.f);
            float v1 = fmaxf(fmaf(v.y, sa.y, sb.y), 0.f);
            float v2 = fmaxf(fmaf(v.z, sa.z, sb.z), 0.f);
            float v3 = fmaxf(fmaf(v.w, sa.w, sb.w), 0.f);
            x2 = fmaf(v0, v0, fmaf(v1, v1, fmaf(v2, v2, fmaf(v3, v3, x2))));
            wl[lane * 68 + 2 * i]     = pack2bf(v0, v1);
            wl[lane * 68 + 2 * i + 1] = pack2bf(v2, v3);
            l16[(((ytb + (4 * i + 0) * 20) + tw) * 2) | tp] = f2bf(v0);
            l16[(((ytb + (4 * i + 1) * 20) + tw) * 2) | tp] = f2bf(v1);
            l16[(((ytb + (4 * i + 2) * 20) + tw) * 2) | tp] = f2bf(v2);
            l16[(((ytb + (4 * i + 3) * 20) + tw) * 2) | tp] = f2bf(v3);
        }
        // --- distances + softmax, two 16-token tiles; D: M=k rows, N=token cols ---
        #pragma unroll
        for (int tile = 0; tile < 2; ++tile) {
            const int tbase = tile * 16;
            v8f dacc = {};
            #pragma unroll
            for (int kd = 0; kd < 128; kd += 32) {
                FragBF a, bf;
                const uint4* ca = (const uint4*)(cwA + lh * 64 + (kd >> 1) + hi * 4);
                const uint4* cb = (const uint4*)(cwA + lh * 64 + (kd >> 1) + 8 + hi * 4);
                frag_from(a, ca[0], cb[0]);
                const uint4* xp = (const uint4*)(wl + (tbase + lh) * 68 + (kd >> 1) + hi * 4);
                frag_from(bf, xp[0], xp[2]);
                dacc = wmma_bf16(a, bf, dacc);
            }
            float x2t = __shfl(x2, tbase + lh);
            float sl[8], m = -3.0e38f;
            #pragma unroll
            for (int r = 0; r < 8; ++r) {
                sl[r] = sc[r] * (x2t + cc[r] - 2.f * dacc[r]);
                m = fmaxf(m, sl[r]);
            }
            m = fmaxf(m, __shfl_xor(m, 16));
            float S = 0.f;
            #pragma unroll
            for (int r = 0; r < 8; ++r) { sl[r] = __expf(sl[r] - m); S += sl[r]; }
            S += __shfl_xor(S, 16);
            float inv = 1.f / S;
            const int t2 = tbase + lh, w2 = t2 >> 1, p2 = t2 & 1;
            #pragma unroll
            for (int r = 0; r < 8; ++r) {
                float av = sl[r] * inv;
                asum[r] += av;
                l16[((awb + (r + 8 * hi) * 20 + w2) * 2) | p2] = f2bf(av);
            }
        }
        // --- aggregation: E^T(d,k) += X^T(d,tok) @ A(tok,k) over this 32-token chunk ---
        #pragma unroll
        for (int s = 0; s < 8; ++s) {
            FragBF a, bf;
            const uint4* pa0 = (const uint4*)(wl + ytb + (s * 16 + lh) * 20 + hi * 4);
            const uint4* pa1 = (const uint4*)(wl + ytb + (s * 16 + lh) * 20 + 8 + hi * 4);
            frag_from(a, pa0[0], pa1[0]);
            const uint4* pb = (const uint4*)(wl + awb + lh * 20 + hi * 4);
            frag_from(bf, pb[0], pb[2]);
            Eacc[s] = wmma_bf16(a, bf, Eacc[s]);
        }
    }
    // commit: lane holds k=lh, d = s*16 + 8*hi + r
    #pragma unroll
    for (int s = 0; s < 8; ++s)
        #pragma unroll
        for (int r = 0; r < 8; ++r)
            atomicAdd(&Ews[(size_t)b * 2048 + lh * 128 + s * 16 + 8 * hi + r], Eacc[s][r]);
    #pragma unroll
    for (int r = 0; r < 8; ++r) {
        float v = asum[r];
        v += __shfl_xor(v, 1); v += __shfl_xor(v, 2); v += __shfl_xor(v, 4); v += __shfl_xor(v, 8);
        if (lh == 0) atomicAdd(&asumws[b * 16 + 8 * hi + r], v);
    }
}

// ================= K4a: e = E - asum*C, L2 normalize, pack bf16 words b*1024+dk2 =================
__global__ __launch_bounds__(256) void k4a_norm(
    const float* __restrict__ Ews, const float* __restrict__ asumws,
    const float* __restrict__ codewords, unsigned int* __restrict__ eA)
{
    __shared__ float red[256];
    const int b = blockIdx.x, tid = threadIdx.x;
    float v[8]; float ss = 0.f;
    const int base = tid * 8;
    #pragma unroll
    for (int j = 0; j < 8; ++j) {
        int i = base + j, k = i >> 7;
        float e = Ews[(size_t)b * 2048 + i] - asumws[b * 16 + k] * codewords[i];
        v[j] = e; ss = fmaf(e, e, ss);
    }
    red[tid] = ss; __syncthreads();
    for (int off = 128; off > 0; off >>= 1) {
        if (tid < off) red[tid] += red[tid + off];
        __syncthreads();
    }
    float invn = 1.f / fmaxf(sqrtf(red[0]), 1e-12f);
    #pragma unroll
    for (int j = 0; j < 4; ++j)
        eA[b * 1024 + (base >> 1) + j] = pack2bf(v[2 * j] * invn, v[2 * j + 1] * invn);
}

// ================= K4b: gate = sigmoid(e @ lin_w^T + lin_b), M=16 batches ========================
__global__ __launch_bounds__(256) void k4b_gate(
    const unsigned int* __restrict__ eA, const unsigned int* __restrict__ lwB,
    const float* __restrict__ lin_b, float* __restrict__ gate)
{
    const int tid = threadIdx.x, wave = tid >> 5, lane = tid & 31;
    const int lh = lane & 15, hi = lane >> 4;
    const int ctile = (blockIdx.x * 8 + wave) * 16;
    v8f acc = {};
    for (int kc = 0; kc < 2048; kc += 32) {
        FragBF a, bf;
        const uint4* pa0 = (const uint4*)(eA + lh * 1024 + (kc >> 1) + hi * 4);
        const uint4* pa1 = (const uint4*)(eA + lh * 1024 + (kc >> 1) + 8 + hi * 4);
        frag_from(a, pa0[0], pa1[0]);
        const uint4* pb = (const uint4*)(lwB + (size_t)(ctile + lh) * 1024 + (kc >> 1) + hi * 4);
        frag_from(bf, pb[0], pb[2]);
        acc = wmma_bf16(a, bf, acc);
    }
    float lb = lin_b[ctile + lh];
    #pragma unroll
    for (int r = 0; r < 8; ++r) {
        float z = acc[r] + lb;
        gate[(size_t)(r + 8 * hi) * 512 + ctile + lh] = 1.f / (1.f + __expf(-z));
    }
}

// ================= K5: out = x * gate[b,c] =======================================================
__global__ __launch_bounds__(256) void k5_scale(
    const float* __restrict__ x, const float* __restrict__ gate, float* __restrict__ out)
{
    size_t i = (size_t)blockIdx.x * 256 + threadIdx.x;
    float4 v = ((const float4*)x)[i];
    float g = gate[i >> 10];                // 1024 float4 per (b,c) plane
    float4 o = {v.x * g, v.y * g, v.z * g, v.w * g};
    ((float4*)out)[i] = o;
}

extern "C" void kernel_launch(void* const* d_in, const int* in_sizes, int n_in,
                              void* d_out, int out_size, void* d_ws, size_t ws_size,
                              hipStream_t stream)
{
    (void)in_sizes; (void)n_in; (void)out_size; (void)ws_size;
    const float* x     = (const float*)d_in[0];
    const float* convw = (const float*)d_in[1];
    const float* convb = (const float*)d_in[2];
    const float* bng   = (const float*)d_in[3];
    const float* bnb   = (const float*)d_in[4];
    const float* cwf   = (const float*)d_in[5];
    const float* scl   = (const float*)d_in[6];
    const float* linw  = (const float*)d_in[7];
    const float* linb  = (const float*)d_in[8];
    float* out = (float*)d_out;
    char* w = (char*)d_ws;

    float*          y     = (float*)(w + OFF_Y);
    unsigned short* cwbf  = (unsigned short*)(w + OFF_CWBF);
    unsigned int*   cwA   = (unsigned int*)(w + OFF_CWA);
    float*          c2    = (float*)(w + OFF_C2);
    unsigned int*   lwB   = (unsigned int*)(w + OFF_LWB);
    float*          stats = (float*)(w + OFF_STATS);
    float*          bnp   = (float*)(w + OFF_BNP);
    float*          E     = (float*)(w + OFF_E);
    float*          asum  = (float*)(w + OFF_ASUM);
    unsigned int*   eA    = (unsigned int*)(w + OFF_EA);
    float*          gate  = (float*)(w + OFF_GATE);

    k0_prep  <<<2438, 256, 0, stream>>>(linw, convw, cwf, lwB, cwbf, cwA, c2, stats, E, asum);
    k1_conv  <<<1024, 256, 0, stream>>>(x, cwbf, convb, y, stats);
    k2_bn    <<<1, 128, 0, stream>>>(stats, bng, bnb, bnp);
    k3_encode<<<512, 64, 0, stream>>>(y, bnp, cwA, c2, scl, E, asum);
    k4a_norm <<<16, 256, 0, stream>>>(E, asum, cwf, eA);
    k4b_gate <<<4, 256, 0, stream>>>(eA, lwB, linb, gate);
    k5_scale <<<32768, 256, 0, stream>>>(x, gate, out);
}
